// ExcelFormerConv_81673098101064
// MI455X (gfx1250) — matrix operand order repaired and checked by reading, not compiled, not run on hardware
//
#include <hip/hip_runtime.h>
#include <hip/hip_bf16.h>
#include <math.h>

// MI455X (gfx1250) implementation. All five GEMM-like stages are routed through
// v_wmma_f32_16x16x32_f16 (f16 A/B, f32 accum). Roofline: ~21.5 GFLOP of matmul
// vs ~100 MB of HBM traffic -> WMMA throughput bound; f16 operands halve traffic
// and hit the matrix pipe. Weights are pre-transposed to [N][K] f16 so B-fragments
// are contiguous; v is stored transposed so the PV WMMA's B-fragment is contiguous.

typedef __attribute__((ext_vector_type(16))) _Float16 v16h;
typedef __attribute__((ext_vector_type(8)))  _Float16 v8h;
typedef __attribute__((ext_vector_type(8)))  float    v8f;

union F16x16 { v16h v; v8h h[2]; };

#define DEV static __device__ __forceinline__

DEV v8f wmma_f16(v16h a, v16h b, v8f c) {
  // (neg_a, A, neg_b, B, c_mod, C, reuse_a, reuse_b)
  return __builtin_amdgcn_wmma_f32_16x16x32_f16(false, a, false, b, (short)0, c,
                                                false, false);
}

// A-fragment (16x32, f16, row-major source, leading dim ldm in halves).
// ISA layout: lane L holds row L&15; v16h elems 0..7 = K = 8*(L>>4)..+7,
// elems 8..15 = that +16.  -> two contiguous 16-byte loads per lane.
DEV v16h load_a_frag(const _Float16* base, int ldm, int row0, int k0, int lane) {
  int row = row0 + (lane & 15);
  int kb  = (lane >> 4) << 3;
  const _Float16* p = base + (size_t)row * ldm + k0 + kb;
  F16x16 u;
  u.h[0] = *(const v8h*)(p);
  u.h[1] = *(const v8h*)(p + 16);
  return u.v;
}

// B-fragment (32x16, f16). Source is W^T laid out [N][K] (ldk halves per row)
// so B[k][n] = wt[n][k]. ISA layout: column = lane&15; v16h elems 0..7 = K = 8*(L>>4)..+7,
// elems 8..15 = that +16. -> two contiguous 16-byte loads per lane.
DEV v16h load_b_fragT(const _Float16* wt, int ldk, int n0, int k0, int lane) {
  int n  = n0 + (lane & 15);
  int kb = (lane >> 4) << 3;
  const _Float16* p = wt + (size_t)n * ldk + k0 + kb;
  F16x16 u;
  u.h[0] = *(const v8h*)(p);
  u.h[1] = *(const v8h*)(p + 16);
  return u.v;
}

// ---------------------------------------------------------------- prep: W -> W^T f16
__global__ void transpose_cvt_kernel(_Float16* __restrict__ dst,
                                     const float* __restrict__ src, int K, int N) {
  int idx = blockIdx.x * blockDim.x + threadIdx.x;
  if (idx >= K * N) return;
  int k = idx / N, n = idx - k * N;
  dst[(size_t)n * K + k] = (_Float16)src[idx];
}

// ---------------------------------------------------------------- LayerNorm -> f16
__global__ __launch_bounds__(256) void layernorm_kernel(
    const float* __restrict__ x, const float* __restrict__ gamma,
    const float* __restrict__ beta, _Float16* __restrict__ h) {
  int lane = threadIdx.x & 31;
  int wave = threadIdx.x >> 5;
  int row  = (blockIdx.x << 3) + wave;          // 8 waves/block, one row per wave
  const float* xr = x + (size_t)row * 256;
  const float4* xp = (const float4*)(xr + (lane << 3));
  float4 u0 = xp[0], u1 = xp[1];
  float s = u0.x + u0.y + u0.z + u0.w + u1.x + u1.y + u1.z + u1.w;
  for (int m = 1; m < 32; m <<= 1) s += __shfl_xor(s, m, 32);
  float mu = s * (1.0f / 256.0f);
  float d[8] = {u0.x - mu, u0.y - mu, u0.z - mu, u0.w - mu,
                u1.x - mu, u1.y - mu, u1.z - mu, u1.w - mu};
  float sq = 0.0f;
#pragma unroll
  for (int i = 0; i < 8; ++i) sq += d[i] * d[i];
  for (int m = 1; m < 32; m <<= 1) sq += __shfl_xor(sq, m, 32);
  float rstd = rsqrtf(sq * (1.0f / 256.0f) + 1e-5f);
  const float4* gp = (const float4*)(gamma + (lane << 3));
  const float4* bp = (const float4*)(beta + (lane << 3));
  float4 g0 = gp[0], g1 = gp[1], b0 = bp[0], b1 = bp[1];
  float gg[8] = {g0.x, g0.y, g0.z, g0.w, g1.x, g1.y, g1.z, g1.w};
  float bb[8] = {b0.x, b0.y, b0.z, b0.w, b1.x, b1.y, b1.z, b1.w};
  v8h out;
#pragma unroll
  for (int i = 0; i < 8; ++i) out[i] = (_Float16)(d[i] * rstd * gg[i] + bb[i]);
  *(v8h*)(h + (size_t)row * 256 + (lane << 3)) = out;
}

// ---------------------------------------------------------------- QKV GEMM (WMMA)
// C[16384][768] = h @ w_qkv ; scatter into q,k [bh][s][32] and v^T [bh][32][s].
__global__ __launch_bounds__(128) void qkv_gemm_kernel(
    const _Float16* __restrict__ h, const _Float16* __restrict__ wqkvT,
    _Float16* __restrict__ q, _Float16* __restrict__ k, _Float16* __restrict__ vT) {
  int lane = threadIdx.x & 31;
  int wid  = (blockIdx.x << 2) + (threadIdx.x >> 5);
  int mt = wid / 12, nt = wid - mt * 12;        // 1024 M-tiles x 12 N64-tiles
  int m0 = mt << 4, n0 = nt << 6;
  v8f zero = {0, 0, 0, 0, 0, 0, 0, 0};
  v8f c[4] = {zero, zero, zero, zero};
  for (int k0 = 0; k0 < 256; k0 += 32) {
    v16h a = load_a_frag(h, 256, m0, k0, lane);
#pragma unroll
    for (int t = 0; t < 4; ++t) {
      v16h b = load_b_fragT(wqkvT, 256, n0 + (t << 4), k0, lane);
      c[t] = wmma_f16(a, b, c[t]);
    }
  }
  int halfid = lane >> 4, nc = lane & 15;
#pragma unroll
  for (int t = 0; t < 4; ++t) {
    int ncol = n0 + (t << 4) + nc;
    int sel = ncol >> 8, within = ncol & 255;
    int head = within >> 5, dd = within & 31;
#pragma unroll
    for (int i = 0; i < 8; ++i) {
      int row = m0 + i + (halfid << 3);
      int b_ = row >> 9, s_ = row & 511;
      size_t bh = (size_t)((b_ << 3) + head);
      _Float16 val = (_Float16)c[t][i];
      if (sel == 0)      q [(bh * 512 + s_) * 32 + dd] = val;
      else if (sel == 1) k [(bh * 512 + s_) * 32 + dd] = val;
      else               vT[(bh * 32 + dd) * 512 + s_] = val;
    }
  }
}

// ---------------------------------------------------------------- flash attention
// 1 wave = 16 query rows; block = 4 waves = 64 rows of one (b,h). Causal.
__global__ __launch_bounds__(128) void attn_kernel(
    const _Float16* __restrict__ q, const _Float16* __restrict__ kbuf,
    const _Float16* __restrict__ vT, _Float16* __restrict__ ao) {
  __shared__ _Float16 ldsP[4][16][32];          // per-wave P staging (C->A relayout)
  int lane = threadIdx.x & 31;
  int wave = threadIdx.x >> 5;
  int bh = blockIdx.x >> 3;
  int i0 = ((blockIdx.x & 7) << 6) + (wave << 4);
  int bb = bh >> 3, hh = bh & 7;
  const _Float16* qb = q    + (size_t)bh * 512 * 32;
  const _Float16* kb = kbuf + (size_t)bh * 512 * 32;
  const _Float16* vb = vT   + (size_t)bh * 32 * 512;
  int halfid = lane >> 4, nc = lane & 15;
  const float SCALE = 0.17677669529663687f;     // 1/sqrt(32)
  v8f zero = {0, 0, 0, 0, 0, 0, 0, 0};
  v8f o0 = zero, o1 = zero;
  float m[8], l[8];
#pragma unroll
  for (int i = 0; i < 8; ++i) { m[i] = -3.0e38f; l[i] = 0.0f; }
  v16h qa = load_a_frag(qb, 32, i0, 0, lane);   // K = full head dim (32)
  int ilimit = i0 + 15;
  for (int j0 = 0; j0 <= ilimit; j0 += 32) {
    float s0[8], s1[8];
    {
      v16h bf = load_b_fragT(kb, 32, j0, 0, lane);      // k rows as B^T
      v8f sc = wmma_f16(qa, bf, zero);
#pragma unroll
      for (int i = 0; i < 8; ++i) {
        int row = i0 + i + (halfid << 3);
        int col = j0 + nc;
        s0[i] = ((col <= row) ? sc[i] : sc[i] - 10000.0f) * SCALE;
      }
    }
    bool have1 = (j0 + 16) <= ilimit;                   // wave-uniform
    if (have1) {
      v16h bf = load_b_fragT(kb, 32, j0 + 16, 0, lane);
      v8f sc = wmma_f16(qa, bf, zero);
#pragma unroll
      for (int i = 0; i < 8; ++i) {
        int row = i0 + i + (halfid << 3);
        int col = j0 + 16 + nc;
        s1[i] = ((col <= row) ? sc[i] : sc[i] - 10000.0f) * SCALE;
      }
    } else {
#pragma unroll
      for (int i = 0; i < 8; ++i) s1[i] = -3.0e38f;
    }
    // online softmax; rows live in 16-lane halves -> butterfly within half
    float p0[8], p1[8], alpha[8];
#pragma unroll
    for (int i = 0; i < 8; ++i) {
      float mx = fmaxf(s0[i], s1[i]);
      for (int sh = 1; sh < 16; sh <<= 1) mx = fmaxf(mx, __shfl_xor(mx, sh, 32));
      float mn = fmaxf(m[i], mx);
      alpha[i] = __expf(m[i] - mn);
      m[i] = mn;
      p0[i] = __expf(s0[i] - mn);
      p1[i] = __expf(s1[i] - mn);
      float rs = p0[i] + p1[i];
      for (int sh = 1; sh < 16; sh <<= 1) rs += __shfl_xor(rs, sh, 32);
      l[i] = l[i] * alpha[i] + rs;
    }
#pragma unroll
    for (int i = 0; i < 8; ++i) { o0[i] *= alpha[i]; o1[i] *= alpha[i]; }
    // C-layout -> A-layout relayout of P through per-wave LDS (in-order per wave)
#pragma unroll
    for (int i = 0; i < 8; ++i) {
      ldsP[wave][i + (halfid << 3)][nc]      = (_Float16)p0[i];
      ldsP[wave][i + (halfid << 3)][16 + nc] = (_Float16)p1[i];
    }
    asm volatile("s_wait_dscnt 0" ::: "memory");
    F16x16 pa;
    int ar = lane & 15, akb = (lane >> 4) << 3;
    pa.h[0] = *(const v8h*)&ldsP[wave][ar][akb];
    pa.h[1] = *(const v8h*)&ldsP[wave][ar][akb + 16];
    asm volatile("s_wait_dscnt 0" ::: "memory");
    // P(16x32) @ V(32x32): v^T rows are contiguous B-fragments
    v16h vb0 = load_b_fragT(vb, 512, 0,  j0, lane);     // d = 0..15
    v16h vb1 = load_b_fragT(vb, 512, 16, j0, lane);     // d = 16..31
    o0 = wmma_f16(pa.v, vb0, o0);
    o1 = wmma_f16(pa.v, vb1, o1);
  }
#pragma unroll
  for (int i = 0; i < 8; ++i) {
    int row = i0 + i + (halfid << 3);
    size_t tok = (size_t)bb * 512 + row;
    float inv = 1.0f / l[i];
    ao[tok * 256 + hh * 32 + nc]      = (_Float16)(o0[i] * inv);
    ao[tok * 256 + hh * 32 + 16 + nc] = (_Float16)(o1[i] * inv);
  }
}

// ---------------------------------------------------------------- out proj + bias
__global__ __launch_bounds__(128) void outproj_kernel(
    const _Float16* __restrict__ ao, const _Float16* __restrict__ woutT,
    const float* __restrict__ bout, float* __restrict__ out_f32,
    _Float16* __restrict__ out_f16) {
  int lane = threadIdx.x & 31;
  int wid  = (blockIdx.x << 2) + (threadIdx.x >> 5);
  int mt = wid >> 2, nt = wid & 3;
  int m0 = mt << 4, n0 = nt << 6;
  v8f zero = {0, 0, 0, 0, 0, 0, 0, 0};
  v8f c[4] = {zero, zero, zero, zero};
  for (int k0 = 0; k0 < 256; k0 += 32) {
    v16h a = load_a_frag(ao, 256, m0, k0, lane);
#pragma unroll
    for (int t = 0; t < 4; ++t) {
      v16h b = load_b_fragT(woutT, 256, n0 + (t << 4), k0, lane);
      c[t] = wmma_f16(a, b, c[t]);
    }
  }
  int halfid = lane >> 4, nc = lane & 15;
#pragma unroll
  for (int t = 0; t < 4; ++t) {
    int ncol = n0 + (t << 4) + nc;
    float bv = bout[ncol];
#pragma unroll
    for (int i = 0; i < 8; ++i) {
      int row = m0 + i + (halfid << 3);
      float val = c[t][i] + bv;
      out_f32[(size_t)row * 256 + ncol] = val;
      out_f16[(size_t)row * 256 + ncol] = (_Float16)val;
    }
  }
}

// ---------------------------------------------------------------- GLU epilogue
// y = out + (out @ w_glu)[:,:256] * tanh((out @ w_glu)[:,256:])
__global__ __launch_bounds__(128) void glu_kernel(
    const _Float16* __restrict__ out_f16, const float* __restrict__ out_f32,
    const _Float16* __restrict__ wgluT, float* __restrict__ y) {
  int lane = threadIdx.x & 31;
  int wid  = (blockIdx.x << 2) + (threadIdx.x >> 5);
  int mt = wid >> 4, nt = wid & 15;
  int m0 = mt << 4, n0 = nt << 4;
  v8f zero = {0, 0, 0, 0, 0, 0, 0, 0};
  v8f ca = zero, cg = zero;
  for (int k0 = 0; k0 < 256; k0 += 32) {
    v16h a  = load_a_frag(out_f16, 256, m0, k0, lane);
    v16h bA = load_b_fragT(wgluT, 256, n0,       k0, lane);
    v16h bG = load_b_fragT(wgluT, 256, 256 + n0, k0, lane);
    ca = wmma_f16(a, bA, ca);
    cg = wmma_f16(a, bG, cg);
  }
  int halfid = lane >> 4, nc = lane & 15;
  int ncol = n0 + nc;
#pragma unroll
  for (int i = 0; i < 8; ++i) {
    int row = m0 + i + (halfid << 3);
    float r = out_f32[(size_t)row * 256 + ncol] + ca[i] * tanhf(cg[i]);
    y[(size_t)row * 256 + ncol] = r;
  }
}

// ---------------------------------------------------------------- host launcher
extern "C" void kernel_launch(void* const* d_in, const int* in_sizes, int n_in,
                              void* d_out, int out_size, void* d_ws, size_t ws_size,
                              hipStream_t stream) {
  (void)in_sizes; (void)n_in; (void)out_size; (void)ws_size;
  const float* x        = (const float*)d_in[0];
  const float* ln_gamma = (const float*)d_in[1];
  const float* ln_beta  = (const float*)d_in[2];
  const float* w_qkv    = (const float*)d_in[3];
  const float* w_out    = (const float*)d_in[4];
  const float* b_out    = (const float*)d_in[5];
  const float* w_glu    = (const float*)d_in[6];
  float* y = (float*)d_out;

  // workspace carve-up (all 256-B aligned); total ~64.8 MB
  char* ws = (char*)d_ws;
  size_t off = 0;
  auto take = [&](size_t bytes) { char* p = ws + off; off += (bytes + 255) & ~(size_t)255; return p; };
  _Float16* h_f16   = (_Float16*)take((size_t)16384 * 256 * 2);
  _Float16* wqkvT   = (_Float16*)take((size_t)768 * 256 * 2);
  _Float16* woutT   = (_Float16*)take((size_t)256 * 256 * 2);
  _Float16* wgluT   = (_Float16*)take((size_t)512 * 256 * 2);
  _Float16* qbuf    = (_Float16*)take((size_t)16384 * 256 * 2);
  _Float16* kbuf    = (_Float16*)take((size_t)16384 * 256 * 2);
  _Float16* vTbuf   = (_Float16*)take((size_t)16384 * 256 * 2);
  _Float16* aobuf   = (_Float16*)take((size_t)16384 * 256 * 2);
  float*    out_f32 = (float*)   take((size_t)16384 * 256 * 4);
  _Float16* out_f16 = (_Float16*)take((size_t)16384 * 256 * 2);

  // weight transposes -> f16 [N][K]
  transpose_cvt_kernel<<<(256 * 768 + 255) / 256, 256, 0, stream>>>(wqkvT, w_qkv, 256, 768);
  transpose_cvt_kernel<<<(256 * 256 + 255) / 256, 256, 0, stream>>>(woutT, w_out, 256, 256);
  transpose_cvt_kernel<<<(256 * 512 + 255) / 256, 256, 0, stream>>>(wgluT, w_glu, 256, 512);

  // LayerNorm: 16384 rows, 8 rows/block
  layernorm_kernel<<<2048, 256, 0, stream>>>(x, ln_gamma, ln_beta, h_f16);

  // QKV: 1024 M-tiles x 12 N64-tiles = 12288 wave-tiles / 4 waves
  qkv_gemm_kernel<<<3072, 128, 0, stream>>>(h_f16, wqkvT, qbuf, kbuf, vTbuf);

  // attention: 256 (b,h) x 8 blocks of 64 query rows
  attn_kernel<<<2048, 128, 0, stream>>>(qbuf, kbuf, vTbuf, aobuf);

  // out projection: 1024 M-tiles x 4 N64-tiles = 4096 wave-tiles / 4 waves
  outproj_kernel<<<1024, 128, 0, stream>>>(aobuf, woutT, b_out, out_f32, out_f16);

  // GLU: 1024 M-tiles x 16 N-tiles = 16384 wave-tiles / 4 waves
  glu_kernel<<<4096, 128, 0, stream>>>(out_f16, out_f32, wgluT, y);
}
